// OriginalLSTM_53566832115845
// MI455X (gfx1250) — matrix operand order repaired and checked by reading, not compiled, not run on hardware
//
#include <hip/hip_runtime.h>

typedef __attribute__((ext_vector_type(16))) __bf16 v16bf;
typedef __attribute__((ext_vector_type(8)))  __bf16 v8bf;
typedef __attribute__((ext_vector_type(8)))  float  v8f;

#define WMMA_BF16(Am, Bm, Cm) \
  __builtin_amdgcn_wmma_f32_16x16x32_bf16(false, (Am), false, (Bm), (short)0, (Cm), false, false)

namespace {
constexpr int kB  = 2048;   // batch
constexpr int kT  = 512;    // seq len
constexpr int kD  = 5;      // input dim
constexpr int kH  = 128;    // hidden
constexpr int kC  = 10;     // classes
constexpr int kG  = 4 * kH; // 512 gate columns
constexpr int MT  = 16;     // batch rows per block (1 WMMA M-tile)
constexpr int NTH = 256;    // 8 waves of 32 -> 2 waves/SIMD -> ~512 VGPR/wave budget
constexpr int NTW = 4;      // N-tiles (of 16 cols) owned per wave: 8 waves * 64 cols = 512
constexpr int GP  = kG + 4; // padded f32 gate stride (bank-conflict pad)
constexpr int HP  = kH + 8; // padded bf16 hidden stride (16B pad)
}

__device__ __forceinline__ float sig_(float x) { return 1.0f / (1.0f + __expf(-x)); }
__device__ __forceinline__ float tanh_(float x) {
  float e = __expf(2.0f * x);
  return (e - 1.0f) / (e + 1.0f);
}

// B fragment (K=32 x N=16, bf16): lane holds column n; lanes 0-15 carry K=kb*32+[0..15],
// lanes 16-31 carry K=kb*32+[16..31]; vector elements are consecutive K.
// Source is a row-major f32 weight [4H][H] (row = gate column), converted to bf16 once.
__device__ __forceinline__ v16bf load_b_frag(const float* __restrict__ w, int ncol, int kb, int hi16) {
  const float* s = w + ncol * kH + kb * 32 + hi16 * 16;
  v16bf f;
#pragma unroll
  for (int i = 0; i < 16; ++i) f[i] = (__bf16)s[i];
  return f;
}

// A fragment (M=16 x K=32, bf16) from LDS row-major [MT][HP]:
// lane row m = lane&15; lanes 0-15: K = kb*32 + [0..15]; lanes 16-31: K = kb*32 + [16..31].
__device__ __forceinline__ v16bf load_a_frag(const __bf16* __restrict__ hbuf, int kb,
                                             int l16, int hi16) {
  const __bf16* s = hbuf + l16 * HP + kb * 32 + hi16 * 16;
  v8bf lo = *(const v8bf*)(s);
  v8bf hi = *(const v8bf*)(s + 8);
  return __builtin_shufflevector(lo, hi, 0, 1, 2, 3, 4, 5, 6, 7, 8, 9, 10, 11, 12, 13, 14, 15);
}

__global__ __launch_bounds__(NTH) void lstm2_fused_kernel(
    const float* __restrict__ x,
    const float* __restrict__ w_ih0, const float* __restrict__ w_hh0,
    const float* __restrict__ b_ih0, const float* __restrict__ b_hh0,
    const float* __restrict__ w_ih1, const float* __restrict__ w_hh1,
    const float* __restrict__ b_ih1, const float* __restrict__ b_hh1,
    const float* __restrict__ cw1, const float* __restrict__ cb1,
    const float* __restrict__ cw2, const float* __restrict__ cb2,
    float* __restrict__ out) {
  __shared__ __align__(16) float  gates[MT * GP];     // 33 KB, reused by both layers
  __shared__ __align__(16) __bf16 h0buf[MT * HP];     // layer-0 hidden state (bf16)
  __shared__ __align__(16) __bf16 h1buf[MT * HP];     // layer-1 hidden state (bf16)
  __shared__ __align__(16) float  c0buf[MT * kH];     // layer-0 cell state (f32, in LDS
  __shared__ __align__(16) float  c1buf[MT * kH];     //  to keep VGPRs for weights)
  __shared__ __align__(16) float  h1f32[MT * kH];     // full-precision final hidden
  __shared__ __align__(16) float  xbuf[MT * 8];       // x_t tile
  __shared__ __align__(16) float  wih0s[kG * kD];     // input projection weights (f32)
  __shared__ __align__(16) float  b0s[kG];
  __shared__ __align__(16) float  b1s[kG];
  __shared__ __align__(16) float  rbuf[MT * 64];      // classifier hidden

  const int tid  = threadIdx.x;
  const int lane = tid & 31;
  const int wave = tid >> 5;     // 0..7, owns gate columns [wave*64, wave*64+64)
  const int l16  = lane & 15;
  const int hi16 = lane >> 4;
  const int row0 = blockIdx.x * MT;

  // ---- one-time staging ----
  for (int e = tid; e < kG * kD; e += NTH) wih0s[e] = w_ih0[e];
  for (int e = tid; e < kG; e += NTH) {
    b0s[e] = b_ih0[e] + b_hh0[e];
    b1s[e] = b_ih1[e] + b_hh1[e];
  }
  for (int e = tid; e < MT * HP; e += NTH) {
    h0buf[e] = (__bf16)0.0f;
    h1buf[e] = (__bf16)0.0f;
  }
  for (int e = tid; e < MT * kH; e += NTH) {
    c0buf[e] = 0.0f;
    c1buf[e] = 0.0f;
  }

  // Register-resident bf16 weight fragments, reused for all 512 timesteps.
  // 48 fragments * 8 VGPRs = 384 VGPRs/wave; 8 waves * 384 regs * 128B = 384KB/WGP.
  v16bf whh0f[4][NTW], wih1f[4][NTW], whh1f[4][NTW];
#pragma unroll
  for (int kb = 0; kb < 4; ++kb) {
#pragma unroll
    for (int nt = 0; nt < NTW; ++nt) {
      const int ncol = wave * 64 + nt * 16 + l16;
      whh0f[kb][nt] = load_b_frag(w_hh0, ncol, kb, hi16);
      wih1f[kb][nt] = load_b_frag(w_ih1, ncol, kb, hi16);
      whh1f[kb][nt] = load_b_frag(w_hh1, ncol, kb, hi16);
    }
  }

  __syncthreads();

  const v8f Z8 = {0.f, 0.f, 0.f, 0.f, 0.f, 0.f, 0.f, 0.f};
  const int n0 = wave * 64 + l16;

  for (int t = 0; t < kT; ++t) {
    // stage x_t tile (16 rows x 5)
    for (int e = tid; e < MT * kD; e += NTH) {
      const int m = e / kD, d = e % kD;
      xbuf[m * 8 + d] = x[((size_t)(row0 + m) * kT + t) * kD + d];
    }
    __syncthreads();

    // ---- layer 0 recurrent GEMM: gates = h0 @ Whh0^T ----
    {
      v8f acc[NTW] = {Z8, Z8, Z8, Z8};
#pragma unroll
      for (int kb = 0; kb < 4; ++kb) {
        const v16bf p = load_a_frag(h0buf, kb, l16, hi16);
#pragma unroll
        for (int nt = 0; nt < NTW; ++nt)
          acc[nt] = WMMA_BF16(p, whh0f[kb][nt], acc[nt]);
      }
#pragma unroll
      for (int v = 0; v < 8; ++v) {
        const int m0 = hi16 * 8 + v;
#pragma unroll
        for (int nt = 0; nt < NTW; ++nt)
          gates[m0 * GP + n0 + 16 * nt] = acc[nt][v];
      }
    }
    __syncthreads();

    // ---- layer 0 cell update (adds bias + x_t @ Wih0^T on the VALU) ----
#pragma unroll
    for (int j = 0; j < 8; ++j) {
      const int e = tid + j * NTH;          // 0..2047
      const int m = e >> 7, h = e & 127;
      float pre[4];
#pragma unroll
      for (int g = 0; g < 4; ++g) {
        const int col = h + g * kH;
        float s = gates[m * GP + col] + b0s[col];
        const float* wr = &wih0s[col * kD];
#pragma unroll
        for (int d = 0; d < kD; ++d) s += xbuf[m * 8 + d] * wr[d];
        pre[g] = s;
      }
      const float ig = sig_(pre[0]);
      const float fg = sig_(pre[1]);
      const float gg = tanh_(pre[2]);
      const float og = sig_(pre[3]);
      const float c  = fg * c0buf[e] + ig * gg;
      c0buf[e] = c;
      h0buf[m * HP + h] = (__bf16)(og * tanh_(c));
    }
    __syncthreads();

    // ---- layer 1 GEMMs: gates = h0_t @ Wih1^T + h1 @ Whh1^T ----
    {
      v8f acc[NTW] = {Z8, Z8, Z8, Z8};
#pragma unroll
      for (int kb = 0; kb < 4; ++kb) {
        {
          const v16bf p = load_a_frag(h0buf, kb, l16, hi16);
#pragma unroll
          for (int nt = 0; nt < NTW; ++nt)
            acc[nt] = WMMA_BF16(p, wih1f[kb][nt], acc[nt]);
        }
        {
          const v16bf q = load_a_frag(h1buf, kb, l16, hi16);
#pragma unroll
          for (int nt = 0; nt < NTW; ++nt)
            acc[nt] = WMMA_BF16(q, whh1f[kb][nt], acc[nt]);
        }
      }
#pragma unroll
      for (int v = 0; v < 8; ++v) {
        const int m0 = hi16 * 8 + v;
#pragma unroll
        for (int nt = 0; nt < NTW; ++nt)
          gates[m0 * GP + n0 + 16 * nt] = acc[nt][v];
      }
    }
    __syncthreads();

    // ---- layer 1 cell update ----
#pragma unroll
    for (int j = 0; j < 8; ++j) {
      const int e = tid + j * NTH;
      const int m = e >> 7, h = e & 127;
      const float pi = gates[m * GP + h]          + b1s[h];
      const float pf = gates[m * GP + h + kH]     + b1s[h + kH];
      const float pg = gates[m * GP + h + 2 * kH] + b1s[h + 2 * kH];
      const float po = gates[m * GP + h + 3 * kH] + b1s[h + 3 * kH];
      const float c  = sig_(pf) * c1buf[e] + sig_(pi) * tanh_(pg);
      c1buf[e] = c;
      const float hn = sig_(po) * tanh_(c);
      h1buf[m * HP + h] = (__bf16)hn;
      h1f32[e] = hn;            // only the final timestep's value survives the loop
    }
    __syncthreads();
  }

  // ---- write last_hidden (f32, coalesced) ----
  for (int e = tid; e < MT * kH; e += NTH)
    out[(size_t)kB * kC + (size_t)row0 * kH + e] = h1f32[e];

  // ---- fused classifier: relu(h1 @ cw1^T + cb1) @ cw2^T + cb2 ----
  for (int e = tid; e < MT * 64; e += NTH) {
    const int m = e >> 6, j = e & 63;
    float s = cb1[j];
    const float* wr = &cw1[j * kH];
#pragma unroll 8
    for (int k = 0; k < kH; ++k) s += h1f32[m * kH + k] * wr[k];
    rbuf[m * 64 + j] = fmaxf(s, 0.0f);
  }
  __syncthreads();
  for (int e = tid; e < MT * kC; e += NTH) {
    const int m = e / kC, j = e % kC;
    float s = cb2[j];
    const float* wr = &cw2[j * 64];
#pragma unroll 8
    for (int k = 0; k < 64; ++k) s += rbuf[m * 64 + k] * wr[k];
    out[(size_t)(row0 + m) * kC + j] = s;
  }
}

extern "C" void kernel_launch(void* const* d_in, const int* in_sizes, int n_in,
                              void* d_out, int out_size, void* d_ws, size_t ws_size,
                              hipStream_t stream) {
  (void)in_sizes; (void)n_in; (void)out_size; (void)d_ws; (void)ws_size;
  const float* x     = (const float*)d_in[0];
  const float* w_ih0 = (const float*)d_in[1];
  const float* w_hh0 = (const float*)d_in[2];
  const float* b_ih0 = (const float*)d_in[3];
  const float* b_hh0 = (const float*)d_in[4];
  const float* w_ih1 = (const float*)d_in[5];
  const float* w_hh1 = (const float*)d_in[6];
  const float* b_ih1 = (const float*)d_in[7];
  const float* b_hh1 = (const float*)d_in[8];
  const float* cw1   = (const float*)d_in[9];
  const float* cb1   = (const float*)d_in[10];
  const float* cw2   = (const float*)d_in[11];
  const float* cb2   = (const float*)d_in[12];
  float* out = (float*)d_out;

  dim3 grid(kB / MT);   // 128 persistent batch-tile blocks
  dim3 block(NTH);      // 8 waves -> 2 waves/SIMD -> large VGPR budget
  lstm2_fused_kernel<<<grid, block, 0, stream>>>(
      x, w_ih0, w_hh0, b_ih0, b_hh0, w_ih1, w_hh1, b_ih1, b_hh1,
      cw1, cb1, cw2, cb2, out);
}
